// TreeTransformerSparsePostv2_64278480552500
// MI455X (gfx1250) — compile-verified, matrix operand order checked by reading, not run on hardware
//
#include <hip/hip_runtime.h>
#include <hip/hip_bf16.h>

// ---------------------------------------------------------------------------
// TreeTransformerSparsePostv2 for MI455X (gfx1250).
// All GEMM-shaped work (convs-as-tap-GEMMs, projections, attention einsums,
// FFNs) runs through v_wmma_f32_16x16x32_bf16 with f32 accumulation.
// A-tiles are staged with GLOBAL_LOAD_ASYNC_TO_LDS_B128 (ASYNCcnt path).
// ---------------------------------------------------------------------------

typedef __attribute__((ext_vector_type(16))) __bf16 v16bf;
typedef __attribute__((ext_vector_type(8)))  float  v8f;

#define BM 128
#define BN 64
#define BK 32
#define LDSA 40
#define LDSB 40
#define PAD 8
#define TP  528   // 512 + 2*PAD

enum { F_BIAS = 1, F_ACCUM = 2, F_RELU = 4, F_ADDRES = 8, F_STF32 = 16, F_STBF16 = 32 };

struct GemmArgs {
  const __bf16* A; long long sAb; int lda;   // row-major [M,K]
  const __bf16* B; long long sBb; int ldb;   // row-major [K,N]
  const float* bias;                         // [N] or null
  const float* R; long long sRb; int ldr;    // residual (f32) or null
  float* C;  long long sCb;  int ldc;        // f32 out / accumulator
  __bf16* Cb; long long sCbb; int ldcb;      // bf16 out or null
  int M, N, K, flags;
};

__device__ __forceinline__ __bf16 f2bf(float f) {
  unsigned int u = __builtin_bit_cast(unsigned int, f);
  unsigned int r = u + 0x7FFFu + ((u >> 16) & 1u);
  unsigned short h = (unsigned short)(r >> 16);
  return __builtin_bit_cast(__bf16, h);
}

// LDS aperture places the byte offset in addr[31:0] (ISA 10.2), so the low
// 32 bits of a generic pointer to a __shared__ object are the LDS address.
__device__ __forceinline__ unsigned lds_addr32(const void* p) {
  return (unsigned)(unsigned long long)(size_t)p;
}

// ------------------------------- WMMA GEMM --------------------------------
// Block tile 128x64, K-step 32. 8 waves (wave32) in a 4x2 grid; each wave
// computes a 32x32 tile = 2x2 v_wmma_f32_16x16x32_bf16.
// Requires M%128==0, N%64==0, K%32==0 (true for every call site here).
__global__ __launch_bounds__(256) void k_gemm(GemmArgs g) {
  __shared__ __bf16 As[BM][LDSA];   // [row][k], rows 80B apart (16B aligned)
  __shared__ __bf16 Bs[BN][LDSB];   // transposed: [n][k]
  const int tid  = threadIdx.x;
  const int lane = tid & 31;
  const int wave = tid >> 5;
  const int wm = wave >> 1, wn = wave & 1;
  const int bz = blockIdx.z;

  const __bf16* Ab = g.A + (long long)bz * g.sAb + (long long)blockIdx.y * BM * g.lda;
  const __bf16* Bb = g.B + (long long)bz * g.sBb + (long long)blockIdx.x * BN;

  const int ar = tid >> 1, ac = (tid & 1) << 4;   // A: 128 rows x 32 cols, 16 elems/thread
  const int br = tid >> 3, bc = (tid & 7) << 3;   // B: 32 rows x 64 cols, 8 elems/thread
  const int k0  = (lane < 16) ? 0 : 8;            // per-lane K base (ISA A/B layout)
  const int l15 = lane & 15;

  const unsigned ldsA = lds_addr32(&As[ar][ac]);  // same dest every K-step

  v8f acc[2][2] = {};

  for (int kb = 0; kb < g.K; kb += BK) {
    // A tile: direct global->LDS DMA, two 16B chunks (offset hits both sides).
    const __bf16* ap = Ab + (long long)ar * g.lda + (kb + ac);
    const unsigned long long ga = (unsigned long long)(size_t)ap;
    asm volatile("global_load_async_to_lds_b128 %0, %1, off"
                 :: "v"(ldsA), "v"(ga) : "memory");
    asm volatile("global_load_async_to_lds_b128 %0, %1, off offset:16"
                 :: "v"(ldsA), "v"(ga) : "memory");

    // B tile: VGPR staging (we must transpose into LDS anyway).
    const __bf16* bp = Bb + (long long)(kb + br) * g.ldb + bc;
    uint4 bv = *(const uint4*)(bp);
    const __bf16* be = (const __bf16*)&bv;
#pragma unroll
    for (int j = 0; j < 8; ++j) Bs[bc + j][br] = be[j];

    if (kb + BK < g.K) {   // -> global_prefetch_b8 for next tiles
      __builtin_prefetch(ap + BK, 0, 1);
      __builtin_prefetch(bp + (long long)BK * g.ldb, 0, 1);
    }
    asm volatile("s_wait_asynccnt 0x0" ::: "memory");
    __syncthreads();

    v16bf af[2], bfr[2];
#pragma unroll
    for (int i = 0; i < 2; ++i) {
      const int r = wm * 32 + i * 16 + l15;
      ((uint4*)&af[i])[0] = *(const uint4*)(&As[r][k0]);
      ((uint4*)&af[i])[1] = *(const uint4*)(&As[r][k0 + 16]);
    }
#pragma unroll
    for (int j = 0; j < 2; ++j) {
      const int c = wn * 32 + j * 16 + l15;
      ((uint4*)&bfr[j])[0] = *(const uint4*)(&Bs[c][k0]);
      ((uint4*)&bfr[j])[1] = *(const uint4*)(&Bs[c][k0 + 16]);
    }
#pragma unroll
    for (int i = 0; i < 2; ++i)
#pragma unroll
      for (int j = 0; j < 2; ++j)
        acc[i][j] = __builtin_amdgcn_wmma_f32_16x16x32_bf16(
            false, af[i], false, bfr[j], (short)0, acc[i][j], false, false);
    __syncthreads();   // keep next iteration's async stores off live LDS reads
  }

  // C/D layout: lanes 0-15 -> M = vg, lanes 16-31 -> M = vg+8; N = lane%16.
  const int gm0 = blockIdx.y * BM + wm * 32;
  const int gn0 = blockIdx.x * BN + wn * 32;
  const int mb  = (lane < 16) ? 0 : 8;
#pragma unroll
  for (int i = 0; i < 2; ++i) {
#pragma unroll
    for (int j = 0; j < 2; ++j) {
      const int gn = gn0 + j * 16 + l15;
      const float bia = (g.flags & F_BIAS) ? g.bias[gn] : 0.0f;
#pragma unroll
      for (int vg = 0; vg < 8; ++vg) {
        const int gm = gm0 + i * 16 + mb + vg;
        float v = acc[i][j][vg] + bia;
        if (g.flags & F_ACCUM)  v += g.C[(long long)bz * g.sCb + (long long)gm * g.ldc + gn];
        if (g.flags & F_RELU)   v = v > 0.0f ? v : 0.0f;
        if (g.flags & F_ADDRES) v += g.R[(long long)bz * g.sRb + (long long)gm * g.ldr + gn];
        if (g.flags & F_STF32)  g.C[(long long)bz * g.sCb + (long long)gm * g.ldc + gn] = v;
        if (g.flags & F_STBF16) g.Cb[(long long)bz * g.sCbb + (long long)gm * g.ldcb + gn] = f2bf(v);
      }
    }
  }
}

// --------------------------- reductions / softmax --------------------------
__device__ __forceinline__ float wred_max(float v) {
#pragma unroll
  for (int o = 16; o; o >>= 1) v = fmaxf(v, __shfl_xor(v, o, 32));
  return v;
}
__device__ __forceinline__ float wred_sum(float v) {
#pragma unroll
  for (int o = 16; o; o >>= 1) v += __shfl_xor(v, o, 32);
  return v;
}

__global__ __launch_bounds__(256) void k_softmax(const float* S, __bf16* P, int cols, float scale) {
  __shared__ float red[8];
  const long long row = blockIdx.x;
  const float* s = S + row * (long long)cols;
  __bf16* p = P + row * (long long)cols;
  const int lane = threadIdx.x & 31;

  float mx = -3.0e38f;
  for (int c = threadIdx.x; c < cols; c += 256) mx = fmaxf(mx, s[c]);
  mx = wred_max(mx);
  if (lane == 0) red[threadIdx.x >> 5] = mx;
  __syncthreads();
  mx = wred_max((lane < 8) ? red[lane] : -3.0e38f);
  __syncthreads();

  float sum = 0.0f;
  for (int c = threadIdx.x; c < cols; c += 256) sum += __expf((s[c] - mx) * scale);
  sum = wred_sum(sum);
  if (lane == 0) red[threadIdx.x >> 5] = sum;
  __syncthreads();
  sum = wred_sum((lane < 8) ? red[lane] : 0.0f);
  const float rinv = 1.0f / sum;
  for (int c = threadIdx.x; c < cols; c += 256)
    p[c] = f2bf(__expf((s[c] - mx) * scale) * rinv);
}

// Custom LayerNorm: torch.std (ddof=1), denominator (std + eps).
__global__ __launch_bounds__(256) void k_layernorm(const float* X, const float* ga, const float* gb,
                                                   float* Y, __bf16* Yb) {
  __shared__ float red[8];
  const long long t = blockIdx.x;
  const float* x = X + t * 2048;
  const int lane = threadIdx.x & 31;
  float xl[8], s = 0.0f;
#pragma unroll
  for (int k = 0; k < 8; ++k) { xl[k] = x[threadIdx.x + k * 256]; s += xl[k]; }
  s = wred_sum(s);
  if (lane == 0) red[threadIdx.x >> 5] = s;
  __syncthreads();
  const float mean = wred_sum((lane < 8) ? red[lane] : 0.0f) * (1.0f / 2048.0f);
  __syncthreads();
  float q = 0.0f;
#pragma unroll
  for (int k = 0; k < 8; ++k) { const float d = xl[k] - mean; q += d * d; }
  q = wred_sum(q);
  if (lane == 0) red[threadIdx.x >> 5] = q;
  __syncthreads();
  const float var = wred_sum((lane < 8) ? red[lane] : 0.0f) * (1.0f / 2047.0f);
  const float rs = 1.0f / (sqrtf(var) + 1e-6f);
#pragma unroll
  for (int k = 0; k < 8; ++k) {
    const int c = threadIdx.x + k * 256;
    const float y = ga[c] * (xl[k] - mean) * rs + gb[c];
    Y[t * 2048 + c] = y;
    if (Yb) Yb[t * 2048 + c] = f2bf(y);
  }
}

// ------------------------------ elementwise --------------------------------
__global__ __launch_bounds__(256) void k_cvt(const float* x, __bf16* y, long long n) {
  const long long i = (long long)blockIdx.x * 256 + threadIdx.x;
  if (i < n) y[i] = f2bf(x[i]);
}
__global__ __launch_bounds__(256) void k_zero(__bf16* y, long long n) {
  const long long i = (long long)blockIdx.x * 256 + threadIdx.x;
  if (i < n) y[i] = __builtin_bit_cast(__bf16, (unsigned short)0);
}
// torch conv weight [O,I,3] -> per-tap GEMM B matrices bf16 [3][I][O]
__global__ __launch_bounds__(256) void k_prep_convw(const float* w, __bf16* o, int I, int O, long long n) {
  const long long i = (long long)blockIdx.x * 256 + threadIdx.x;
  if (i >= n) return;
  const int oo = (int)(i % O);
  long long r = i / O;
  const int ii = (int)(r % I);
  const int tap = (int)(r / I);
  o[i] = f2bf(w[((long long)oo * I + ii) * 3 + tap]);
}
// motion_raw [B,256,T] f32 -> mpad0 [B][TP][256] bf16 (time-major, zero pad)
__global__ __launch_bounds__(256) void k_pack_motion(const float* m, __bf16* o, long long n) {
  const long long i = (long long)blockIdx.x * 256 + threadIdx.x;
  if (i >= n) return;
  const int c = (int)(i & 255);
  const long long r = i >> 8;
  const int t = (int)(r & 511);
  const int b = (int)(r >> 9);
  o[((long long)b * TP + PAD + t) * 256 + c] = f2bf(m[((long long)b * 256 + c) * 512 + t]);
}
// proj f32 [B,L,2048] -> bf16 [B,8,S=4L,64]   (view(B,-1,8,64).transpose(1,2))
__global__ __launch_bounds__(256) void k_pack_heads(const float* X, __bf16* Y, int L, int S, long long n) {
  const long long i = (long long)blockIdx.x * 256 + threadIdx.x;
  if (i >= n) return;
  const int d = (int)(i & 63);
  long long r = i >> 6;
  const int t = (int)(r % S); r /= S;
  const int h = (int)(r & 7);
  const int b = (int)(r >> 3);
  Y[i] = f2bf(X[((long long)b * L + (t >> 2)) * 2048 + (t & 3) * 512 + h * 64 + d]);
}
// same but K transposed: bf16 [B,8,64,S]
__global__ __launch_bounds__(256) void k_pack_headsT(const float* X, __bf16* Y, int L, int S, long long n) {
  const long long i = (long long)blockIdx.x * 256 + threadIdx.x;
  if (i >= n) return;
  const int t = (int)(i % S);
  long long r = i / S;
  const int d = (int)(r & 63); r >>= 6;
  const int h = (int)(r & 7);
  const int b = (int)(r >> 3);
  Y[i] = f2bf(X[((long long)b * L + (t >> 2)) * 2048 + (t & 3) * 512 + h * 64 + d]);
}
// O f32 [B,8,256,64] -> bf16 [B,64,2048]
__global__ __launch_bounds__(256) void k_unpack_attn(const float* O, __bf16* Y, long long n) {
  const long long i = (long long)blockIdx.x * 256 + threadIdx.x;
  if (i >= n) return;
  const int c = (int)(i & 2047);
  const long long r = i >> 11;
  const int l = (int)(r & 63);
  const int b = (int)(r >> 6);
  const int s2 = c >> 9, h = (c >> 6) & 7, d = c & 63;
  Y[i] = f2bf(O[(((long long)(b * 8 + h)) * 256 + (l * 4 + s2)) * 64 + d]);
}
__global__ __launch_bounds__(256) void k_final(const float* X, const float* pm, const float* al,
                                               const float* pf, float* out, long long n) {
  const long long i = (long long)blockIdx.x * 256 + threadIdx.x;
  if (i >= n) return;
  const long long r = i >> 11;           // token index b*64 + l
  const float a = al[r];
  out[i] = a * (X[i] * pm[r]) + (1.0f - a) * pf[i];
}

// ------------------------------- host side ---------------------------------
static inline void gemm(hipStream_t st,
                        const __bf16* A, long long sAb, int lda,
                        const __bf16* B, long long sBb, int ldb,
                        const float* bias,
                        const float* R, long long sRb, int ldr,
                        float* C, long long sCb, int ldc,
                        __bf16* Cb, long long sCbb, int ldcb,
                        int M, int N, int K, int flags, int bat) {
  GemmArgs g{A, sAb, lda, B, sBb, ldb, bias, R, sRb, ldr, C, sCb, ldc, Cb, sCbb, ldcb, M, N, K, flags};
  dim3 grid((unsigned)(N / BN), (unsigned)(M / BM), (unsigned)bat);
  k_gemm<<<grid, dim3(256), 0, st>>>(g);
}

#define NB(n) ((unsigned)(((n) + 255) / 256))

// Input order: setup_inputs() dict order, params flattened depth-first in
// insertion order (conv{w1,b1,wd0-2,bd0-2,wl,bl}, sa{att{v,k,q,m}{W,b},
// ffn{W1,b1,W2,b2}, ln1{a,b}, ln2{a,b}}, ga{same}).
enum {
  IN_PF = 0, IN_MASK, IN_MOTION, IN_MOMASK, IN_PMASK, IN_ALPHA,
  IN_CW1, IN_CB1, IN_CWD0, IN_CWD1, IN_CWD2, IN_CBD0, IN_CBD1, IN_CBD2, IN_CWL, IN_CBL,
  IN_SA_VW, IN_SA_VB, IN_SA_KW, IN_SA_KB, IN_SA_QW, IN_SA_QB, IN_SA_MW, IN_SA_MB,
  IN_SA_W1, IN_SA_B1, IN_SA_W2, IN_SA_B2, IN_SA_LN1A, IN_SA_LN1B, IN_SA_LN2A, IN_SA_LN2B,
  IN_GA_VW, IN_GA_VB, IN_GA_KW, IN_GA_KB, IN_GA_QW, IN_GA_QB, IN_GA_MW, IN_GA_MB,
  IN_GA_W1, IN_GA_B1, IN_GA_W2, IN_GA_B2, IN_GA_LN1A, IN_GA_LN1B, IN_GA_LN2A, IN_GA_LN2B
};

extern "C" void kernel_launch(void* const* d_in, const int* in_sizes, int n_in,
                              void* d_out, int out_size, void* d_ws, size_t ws_size,
                              hipStream_t stream) {
  (void)in_sizes; (void)n_in; (void)out_size; (void)ws_size;
  const float* pf     = (const float*)d_in[IN_PF];
  const float* motion = (const float*)d_in[IN_MOTION];
  const float* pmask  = (const float*)d_in[IN_PMASK];
  const float* alpha  = (const float*)d_in[IN_ALPHA];
  auto F = [&](int idx) { return (const float*)d_in[idx]; };

  const long long HH = 2048ll * 2048ll;  // 4,194,304 (also B*L*H)

  // ---- workspace arena ----
  char* wsb = (char*)d_ws;
  size_t off = 0;
  auto alloc = [&](size_t bytes) -> char* {
    off = (off + 255) & ~(size_t)255;
    char* p = wsb + off;
    off += bytes;
    return p;
  };
  __bf16* wbf[12];  // sa v,k,q,m,W1,W2, ga v,k,q,m,W1,W2
  for (int i = 0; i < 12; ++i) wbf[i] = (__bf16*)alloc(HH * 2);
  __bf16* cw1 = (__bf16*)alloc(3ll * 256 * 1024 * 2);
  __bf16* cwd[3];
  for (int i = 0; i < 3; ++i) cwd[i] = (__bf16*)alloc(3ll * 1024 * 1024 * 2);
  __bf16* cwl   = (__bf16*)alloc(3ll * 1024 * 2048 * 2);
  __bf16* mpad0 = (__bf16*)alloc(32ll * TP * 256 * 2);
  __bf16* mpadA = (__bf16*)alloc(32ll * TP * 1024 * 2);
  __bf16* mpadB = (__bf16*)alloc(32ll * TP * 1024 * 2);
  float*  convacc = (float*)alloc(32ll * 512 * 2048 * 4);  // also aliased as projf
  float*  convA = convacc;
  float*  convB = convacc + 32ll * 512 * 1024;
  float*  projf = convacc;                                  // reuse after convs
  __bf16* m_bf  = (__bf16*)alloc(32ll * 512 * 2048 * 2);
  __bf16* x_bf    = (__bf16*)alloc(HH * 2);
  __bf16* attn_bf = (__bf16*)alloc(HH * 2);
  __bf16* h_bf    = (__bf16*)alloc(HH * 2);
  float*  xa = (float*)alloc(HH * 4);
  float*  xb = (float*)alloc(HH * 4);
  __bf16* Qp = (__bf16*)alloc(32ll * 8 * 256 * 64 * 2);
  __bf16* KT = (__bf16*)alloc(32ll * 8 * 64 * 2048 * 2);
  __bf16* Vp = (__bf16*)alloc(32ll * 8 * 2048 * 64 * 2);
  float*  scores = (float*)alloc(4ll * 8 * 256 * 2048 * 4);   // 67 MB chunk buffer
  __bf16* Pbuf   = (__bf16*)alloc(4ll * 8 * 256 * 2048 * 2);  // 33 MB
  float*  Obuf   = (float*)alloc(32ll * 8 * 256 * 64 * 4);    // 17 MB

  // ---- weight prep (bf16 copies / conv transposes) ----
  const int widx[12] = {IN_SA_VW, IN_SA_KW, IN_SA_QW, IN_SA_MW, IN_SA_W1, IN_SA_W2,
                        IN_GA_VW, IN_GA_KW, IN_GA_QW, IN_GA_MW, IN_GA_W1, IN_GA_W2};
  for (int i = 0; i < 12; ++i)
    k_cvt<<<NB(HH), 256, 0, stream>>>(F(widx[i]), wbf[i], HH);
  {
    long long n = 3ll * 256 * 1024;
    k_prep_convw<<<NB(n), 256, 0, stream>>>(F(IN_CW1), cw1, 256, 1024, n);
    n = 3ll * 1024 * 1024;
    for (int i = 0; i < 3; ++i)
      k_prep_convw<<<NB(n), 256, 0, stream>>>(F(IN_CWD0 + i), cwd[i], 1024, 1024, n);
    n = 3ll * 1024 * 2048;
    k_prep_convw<<<NB(n), 256, 0, stream>>>(F(IN_CWL), cwl, 1024, 2048, n);
  }

  // ---- conv pyramid: each k=3 conv = 3 tap-GEMMs accumulated in f32 ----
  {
    long long n = 32ll * TP * 256;
    k_zero<<<NB(n), 256, 0, stream>>>(mpad0, n);
    n = 32ll * TP * 1024;
    k_zero<<<NB(n), 256, 0, stream>>>(mpadA, n);
    k_zero<<<NB(n), 256, 0, stream>>>(mpadB, n);
    n = 32ll * 512 * 256;
    k_pack_motion<<<NB(n), 256, 0, stream>>>(motion, mpad0, n);
  }
  auto conv_stage = [&](const __bf16* inPad, int Cin, const __bf16* Wt, const float* bias,
                        int Cout, int dil, const float* resid, float* accF,
                        __bf16* outBf, bool paddedOut, bool relu) {
    for (int tap = 0; tap < 3; ++tap) {
      int flags = F_STF32 | ((tap == 0) ? F_BIAS : F_ACCUM);
      const float* R = nullptr; long long sRb = 0; int ldr = 0;
      __bf16* Cb = nullptr; long long sCbb = 0; int ldcb = 0;
      if (tap == 2) {
        if (relu) flags |= F_RELU;
        if (resid) { flags |= F_ADDRES; R = resid; sRb = 512ll * Cout; ldr = Cout; }
        flags |= F_STBF16;
        Cb = paddedOut ? (outBf + (long long)PAD * Cout) : outBf;
        ldcb = Cout;
        sCbb = paddedOut ? (long long)TP * Cout : 512ll * Cout;
      }
      gemm(stream,
           inPad + (long long)(PAD + (tap - 1) * dil) * Cin, (long long)TP * Cin, Cin,
           Wt + (long long)tap * Cin * Cout, 0, Cout,
           bias, R, sRb, ldr,
           accF, 512ll * Cout, Cout,
           Cb, sCbb, ldcb,
           512, Cout, Cin, flags, 32);
    }
  };
  conv_stage(mpad0, 256,  cw1,    F(IN_CB1),  1024, 1, nullptr, convA, mpadA, true,  true);
  conv_stage(mpadA, 1024, cwd[0], F(IN_CBD0), 1024, 2, convA,   convB, mpadB, true,  true);
  conv_stage(mpadB, 1024, cwd[1], F(IN_CBD1), 1024, 4, convB,   convA, mpadA, true,  true);
  conv_stage(mpadA, 1024, cwd[2], F(IN_CBD2), 1024, 8, convA,   convB, mpadB, true,  true);
  conv_stage(mpadB, 1024, cwl,    F(IN_CBL),  2048, 1, nullptr, convacc, m_bf, false, false);

  // ---- SA block ----
  k_cvt<<<NB(HH), 256, 0, stream>>>(pf, x_bf, HH);
  const long long nSA = 32ll * 8 * 256 * 64;
  // q/k/v projections (M = B*L = 2048 tokens)
  gemm(stream, x_bf, 0, 2048, wbf[2], 0, 2048, F(IN_SA_QB), nullptr, 0, 0,
       projf, 0, 2048, nullptr, 0, 0, 2048, 2048, 2048, F_BIAS | F_STF32, 1);
  k_pack_heads<<<NB(nSA), 256, 0, stream>>>(projf, Qp, 64, 256, nSA);
  gemm(stream, x_bf, 0, 2048, wbf[1], 0, 2048, F(IN_SA_KB), nullptr, 0, 0,
       projf, 0, 2048, nullptr, 0, 0, 2048, 2048, 2048, F_BIAS | F_STF32, 1);
  k_pack_headsT<<<NB(nSA), 256, 0, stream>>>(projf, KT, 64, 256, nSA);
  gemm(stream, x_bf, 0, 2048, wbf[0], 0, 2048, F(IN_SA_VB), nullptr, 0, 0,
       projf, 0, 2048, nullptr, 0, 0, 2048, 2048, 2048, F_BIAS | F_STF32, 1);
  k_pack_heads<<<NB(nSA), 256, 0, stream>>>(projf, Vp, 64, 256, nSA);
  // scores [bh:256][256x256], softmax, AV
  gemm(stream, Qp, 256ll * 64, 64, KT, 64ll * 256, 256, nullptr, nullptr, 0, 0,
       scores, 256ll * 256, 256, nullptr, 0, 0, 256, 256, 64, F_STF32, 256);
  k_softmax<<<32 * 8 * 256, 256, 0, stream>>>(scores, Pbuf, 256, 0.125f);
  gemm(stream, Pbuf, 256ll * 256, 256, Vp, 256ll * 64, 64, nullptr, nullptr, 0, 0,
       Obuf, 256ll * 64, 64, nullptr, 0, 0, 256, 64, 256, F_STF32, 256);
  k_unpack_attn<<<NB(HH), 256, 0, stream>>>(Obuf, attn_bf, HH);
  // output projection + residual(program_feat), LN1, FFN, LN2
  gemm(stream, attn_bf, 0, 2048, wbf[3], 0, 2048, F(IN_SA_MB), pf, 0, 2048,
       xa, 0, 2048, nullptr, 0, 0, 2048, 2048, 2048, F_BIAS | F_ADDRES | F_STF32, 1);
  k_layernorm<<<2048, 256, 0, stream>>>(xa, F(IN_SA_LN1A), F(IN_SA_LN1B), xb, x_bf);
  gemm(stream, x_bf, 0, 2048, wbf[4], 0, 2048, F(IN_SA_B1), nullptr, 0, 0,
       xa, 0, 2048, h_bf, 0, 2048, 2048, 2048, 2048, F_BIAS | F_RELU | F_STBF16, 1);
  gemm(stream, h_bf, 0, 2048, wbf[5], 0, 2048, F(IN_SA_B2), xb, 0, 2048,
       xa, 0, 2048, nullptr, 0, 0, 2048, 2048, 2048, F_BIAS | F_ADDRES | F_STF32, 1);
  k_layernorm<<<2048, 256, 0, stream>>>(xa, F(IN_SA_LN2A), F(IN_SA_LN2B), xb, x_bf);

  // ---- GA block (cross-attention to 4T = 2048 motion sub-tokens) ----
  const long long nGA = 32ll * 8 * 2048 * 64;
  gemm(stream, x_bf, 0, 2048, wbf[8], 0, 2048, F(IN_GA_QB), nullptr, 0, 0,
       projf, 0, 2048, nullptr, 0, 0, 2048, 2048, 2048, F_BIAS | F_STF32, 1);
  k_pack_heads<<<NB(nSA), 256, 0, stream>>>(projf, Qp, 64, 256, nSA);
  gemm(stream, m_bf, 0, 2048, wbf[7], 0, 2048, F(IN_GA_KB), nullptr, 0, 0,
       projf, 0, 2048, nullptr, 0, 0, 16384, 2048, 2048, F_BIAS | F_STF32, 1);
  k_pack_headsT<<<NB(nGA), 256, 0, stream>>>(projf, KT, 512, 2048, nGA);
  gemm(stream, m_bf, 0, 2048, wbf[6], 0, 2048, F(IN_GA_VB), nullptr, 0, 0,
       projf, 0, 2048, nullptr, 0, 0, 16384, 2048, 2048, F_BIAS | F_STF32, 1);
  k_pack_heads<<<NB(nGA), 256, 0, stream>>>(projf, Vp, 512, 2048, nGA);
  for (int c = 0; c < 8; ++c) {   // 4 batches (32 bh pairs) per chunk
    const long long bh0 = (long long)c * 32;
    gemm(stream, Qp + bh0 * 256 * 64, 256ll * 64, 64,
         KT + bh0 * 64 * 2048, 64ll * 2048, 2048, nullptr, nullptr, 0, 0,
         scores, 256ll * 2048, 2048, nullptr, 0, 0, 256, 2048, 64, F_STF32, 32);
    k_softmax<<<32 * 256, 256, 0, stream>>>(scores, Pbuf, 2048, 0.125f);
    gemm(stream, Pbuf, 256ll * 2048, 2048,
         Vp + bh0 * 2048 * 64, 2048ll * 64, 64, nullptr, nullptr, 0, 0,
         Obuf + bh0 * 256 * 64, 256ll * 64, 64, nullptr, 0, 0,
         256, 64, 2048, F_STF32, 32);
  }
  k_unpack_attn<<<NB(HH), 256, 0, stream>>>(Obuf, attn_bf, HH);
  gemm(stream, attn_bf, 0, 2048, wbf[9], 0, 2048, F(IN_GA_MB), xb, 0, 2048,
       xa, 0, 2048, nullptr, 0, 0, 2048, 2048, 2048, F_BIAS | F_ADDRES | F_STF32, 1);
  k_layernorm<<<2048, 256, 0, stream>>>(xa, F(IN_GA_LN1A), F(IN_GA_LN1B), xb, x_bf);
  gemm(stream, x_bf, 0, 2048, wbf[10], 0, 2048, F(IN_GA_B1), nullptr, 0, 0,
       xa, 0, 2048, h_bf, 0, 2048, 2048, 2048, 2048, F_BIAS | F_RELU | F_STBF16, 1);
  gemm(stream, h_bf, 0, 2048, wbf[11], 0, 2048, F(IN_GA_B2), xb, 0, 2048,
       xa, 0, 2048, nullptr, 0, 0, 2048, 2048, 2048, F_BIAS | F_ADDRES | F_STF32, 1);
  k_layernorm<<<2048, 256, 0, stream>>>(xa, F(IN_GA_LN2A), F(IN_GA_LN2B), xb, nullptr);

  // ---- final: mask + alpha blend ----
  k_final<<<NB(HH), 256, 0, stream>>>(xb, pmask, alpha, pf, (float*)d_out, HH);
}
